// CrossAttnGate_36086315221252
// MI455X (gfx1250) — compile-verified, matrix operand order checked by reading, not run on hardware
//
#include <hip/hip_runtime.h>
#include <math.h>

typedef __attribute__((ext_vector_type(16))) _Float16 v16h;
typedef __attribute__((ext_vector_type(8)))  float    v8f;

#define DIM   1024
#define NHEAD 16
#define HDIM  64
#define BATCH 8
#define LQ    512
#define LK    1024
#define MQ    (BATCH * LQ)   // 4096
#define MK    (BATCH * LK)   // 8192

// Output modes for the GEMM epilogue
#define OUT_F32  0
#define OUT_F16  1   // f16, row-major [M][N]
#define OUT_F16T 2   // f16, transposed [N][M]

// ---------------------------------------------------------------------------
// Fragment loaders.
// 16-bit A/B fragment layout (CDNA5 ISA 7.12.2): lane%16 = row (A: M-row,
// B: N-col since B is loaded from row-major W[N][K]); lane/16 selects K-half:
//   halves[0..7]  = X[row][k0 + 8*(lane/16) + j]
//   halves[8..15] = X[row][k0 + 16 + 8*(lane/16) + j]
// ---------------------------------------------------------------------------
template<bool SPLIT>
__device__ __forceinline__ v16h load_frag_rm(const float* X0, int ld0,
                                             const float* X1, int ld1, int K1,
                                             int r0, int k0, int lane) {
    int r  = r0 + (lane & 15);
    int kh = (lane >> 4) << 3;
    int ka = k0 + kh;
    int kb = k0 + 16 + kh;
    const float* pa;
    const float* pb;
    if constexpr (SPLIT) {
        pa = (ka < K1) ? (X0 + (size_t)r * ld0 + ka)
                       : (X1 + (size_t)r * ld1 + (ka - K1));
        pb = (kb < K1) ? (X0 + (size_t)r * ld0 + kb)
                       : (X1 + (size_t)r * ld1 + (kb - K1));
    } else {
        pa = X0 + (size_t)r * ld0 + ka;
        pb = X0 + (size_t)r * ld0 + kb;
    }
    float4 f0 = *reinterpret_cast<const float4*>(pa);
    float4 f1 = *reinterpret_cast<const float4*>(pa + 4);
    float4 f2 = *reinterpret_cast<const float4*>(pb);
    float4 f3 = *reinterpret_cast<const float4*>(pb + 4);
    v16h o;
    o[0]=(_Float16)f0.x;  o[1]=(_Float16)f0.y;  o[2]=(_Float16)f0.z;  o[3]=(_Float16)f0.w;
    o[4]=(_Float16)f1.x;  o[5]=(_Float16)f1.y;  o[6]=(_Float16)f1.z;  o[7]=(_Float16)f1.w;
    o[8]=(_Float16)f2.x;  o[9]=(_Float16)f2.y;  o[10]=(_Float16)f2.z; o[11]=(_Float16)f2.w;
    o[12]=(_Float16)f3.x; o[13]=(_Float16)f3.y; o[14]=(_Float16)f3.z; o[15]=(_Float16)f3.w;
    return o;
}

// f16 row-major fragment: two 16B loads, no conversion.
__device__ __forceinline__ v16h load_frag_h(const _Float16* X, int ld,
                                            int r0, int k0, int lane) {
    int r  = r0 + (lane & 15);
    int kh = (lane >> 4) << 3;
    const _Float16* pa = X + (size_t)r * ld + k0 + kh;
    const _Float16* pb = X + (size_t)r * ld + k0 + 16 + kh;
    v16h o;
    float4* op = reinterpret_cast<float4*>(&o);
    op[0] = *reinterpret_cast<const float4*>(pa);
    op[1] = *reinterpret_cast<const float4*>(pb);
    return o;
}

__device__ __forceinline__ float gelu_exact(float x) {
    return 0.5f * x * (1.0f + erff(x * 0.70710678118654752f));
}

// ---------------------------------------------------------------------------
// Generic WMMA GEMM: C[M,N] = act( [A0 ‖ A1](M,K) @ W(N,K)^T + bias )
// One wave -> 32x64 output tile: 2 A-fragments reused across 4 B-fragments,
// 8 WMMAs per k-step.  OUTK selects f32 / f16 / f16-transposed epilogue.
// ---------------------------------------------------------------------------
template<bool SPLIT, int OUTK>
__global__ void __launch_bounds__(256, 1)
gemm_wmma_kernel(const float* __restrict__ A0, int ld0,
                 const float* __restrict__ A1, int ld1, int K1,
                 int M, int N, int K,
                 const float* __restrict__ W,
                 const float* __restrict__ bias,
                 void* __restrict__ Cout, int act) {
    int lane = threadIdx.x & 31;
    int wid  = blockIdx.x * (blockDim.x >> 5) + (threadIdx.x >> 5);
    int tilesM = M >> 5;
    int tilesN = N >> 6;
    if (wid >= tilesM * tilesN) return;
    int mbase = (wid % tilesM) << 5;
    int nbase = (wid / tilesM) << 6;

    v8f acc[2][4] = {};
    for (int k0 = 0; k0 < K; k0 += 32) {
        if (k0 + 32 < K) {
            // prefetch next k-panel of W (64 rows, 2 per lane) -> global_prefetch_b8
            __builtin_prefetch(W + (size_t)(nbase + lane) * K + k0 + 32, 0, 3);
            __builtin_prefetch(W + (size_t)(nbase + 32 + lane) * K + k0 + 32, 0, 3);
        }
        v16h a0 = load_frag_rm<SPLIT>(A0, ld0, A1, ld1, K1, mbase,      k0, lane);
        v16h a1 = load_frag_rm<SPLIT>(A0, ld0, A1, ld1, K1, mbase + 16, k0, lane);
#pragma unroll
        for (int t = 0; t < 4; ++t) {
            v16h bf = load_frag_rm<false>(W, K, W, K, K, nbase + t * 16, k0, lane);
            acc[0][t] = __builtin_amdgcn_wmma_f32_16x16x32_f16(
                false, a0, false, bf, (short)0, acc[0][t], false, false);
            acc[1][t] = __builtin_amdgcn_wmma_f32_16x16x32_f16(
                false, a1, false, bf, (short)0, acc[1][t], false, false);
        }
    }
    int ncol = nbase + (lane & 15);
    int mro  = (lane >> 4) << 3;
#pragma unroll
    for (int s = 0; s < 2; ++s)
#pragma unroll
        for (int t = 0; t < 4; ++t) {
            float bv = bias[ncol + t * 16];
            int n  = ncol + t * 16;
            int m0 = mbase + s * 16 + mro;
            if constexpr (OUTK == OUT_F16T) {
                _Float16 pack[8];
#pragma unroll
                for (int j = 0; j < 8; ++j) pack[j] = (_Float16)(acc[s][t][j] + bv);
                _Float16* CT = (_Float16*)Cout;
                *reinterpret_cast<float4*>(CT + (size_t)n * M + m0) =
                    *reinterpret_cast<float4*>(pack);
            } else {
#pragma unroll
                for (int j = 0; j < 8; ++j) {
                    float v = acc[s][t][j] + bv;
                    if (act == 1) v = gelu_exact(v);
                    if constexpr (OUTK == OUT_F16)
                        ((_Float16*)Cout)[(size_t)(m0 + j) * N + n] = (_Float16)v;
                    else
                        ((float*)Cout)[(size_t)(m0 + j) * N + n] = v;
                }
            }
        }
}

// ---------------------------------------------------------------------------
// Flash attention per (b, h, 16-query tile).  qp/kp are f16 [rows][1024];
// vpT is f16 transposed [feature][MK].  S^T = Kp·Qp^T tiles so each lane owns
// one query column; online softmax (m, Z, T) is lane-local + one shfl_xor(16).
// P D-tiles convert register-locally into the B operand of ctx^T += V^T · P.
// Emits entropy logZ - T/Z per (b,h,q).
// ---------------------------------------------------------------------------
__global__ void __launch_bounds__(256, 1)
attn_kernel(const _Float16* __restrict__ qp,
            const _Float16* __restrict__ kp,
            const _Float16* __restrict__ vpT,
            const unsigned char* __restrict__ kv_mask,
            float* __restrict__ ctx,
            float* __restrict__ ent_bh) {
    int lane = threadIdx.x & 31;
    int wid  = blockIdx.x * (blockDim.x >> 5) + (threadIdx.x >> 5);
    const int qtiles = LQ / 16;                  // 32
    int qtile = wid % qtiles;
    int h     = (wid / qtiles) % NHEAD;
    int b     = wid / (qtiles * NHEAD);
    if (b >= BATCH) return;
    int qbase = qtile * 16;

    const _Float16* Qb  = qp  + (size_t)(b * LQ) * DIM + h * HDIM;
    const _Float16* Kb  = kp  + (size_t)(b * LK) * DIM + h * HDIM;
    const _Float16* VTb = vpT + (size_t)(h * HDIM) * MK + b * LK;  // rows=hd, cols=key
    const unsigned char* mk = kv_mask + (size_t)b * LK;

    // Q fragments as B operands (loop invariant), HD=64 -> two k-steps
    v16h bq0 = load_frag_h(Qb, DIM, qbase, 0, lane);
    v16h bq1 = load_frag_h(Qb, DIM, qbase, 32, lane);

    float mrun = -1e30f, Zrun = 0.0f, Trun = 0.0f;
    v8f cacc[4] = {};

    for (int kc = 0; kc < LK; kc += 32) {
        // S^T tiles: rows = keys (kc..kc+15, kc+16..kc+31), cols = queries
        v8f s0 = {}, s1 = {};
        {
            v16h ak = load_frag_h(Kb, DIM, kc, 0, lane);
            s0 = __builtin_amdgcn_wmma_f32_16x16x32_f16(false, ak, false, bq0, (short)0, s0, false, false);
            ak = load_frag_h(Kb, DIM, kc, 32, lane);
            s0 = __builtin_amdgcn_wmma_f32_16x16x32_f16(false, ak, false, bq1, (short)0, s0, false, false);
        }
        {
            v16h ak = load_frag_h(Kb, DIM, kc + 16, 0, lane);
            s1 = __builtin_amdgcn_wmma_f32_16x16x32_f16(false, ak, false, bq0, (short)0, s1, false, false);
            ak = load_frag_h(Kb, DIM, kc + 16, 32, lane);
            s1 = __builtin_amdgcn_wmma_f32_16x16x32_f16(false, ak, false, bq1, (short)0, s1, false, false);
        }
        // scale + mask; D layout: VGPR j -> key = kc + tile*16 + j + 8*(lane/16)
        int kh = (lane >> 4) << 3;
        float sv0[8], sv1[8];
#pragma unroll
        for (int j = 0; j < 8; ++j) {
            sv0[j] = mk[kc + kh + j]      ? s0[j] * 0.125f : -1e30f;
            sv1[j] = mk[kc + 16 + kh + j] ? s1[j] * 0.125f : -1e30f;
        }
        // chunk max over this query column (pairs of lanes l, l^16)
        float cm = sv0[0];
#pragma unroll
        for (int j = 0; j < 8; ++j) { cm = fmaxf(cm, sv0[j]); cm = fmaxf(cm, sv1[j]); }
        cm = fmaxf(cm, __shfl_xor(cm, 16, 32));
        float mnew  = fmaxf(mrun, cm);
        float alpha = __expf(mrun - mnew);
        float psum = 0.0f, tsum = 0.0f;
        float p0[8], p1[8];
#pragma unroll
        for (int j = 0; j < 8; ++j) {
            float d0 = sv0[j] - mnew; float e0 = __expf(d0);
            float d1 = sv1[j] - mnew; float e1 = __expf(d1);
            p0[j] = e0; p1[j] = e1;
            psum += e0 + e1;
            tsum += e0 * d0 + e1 * d1;
        }
        psum += __shfl_xor(psum, 16, 32);
        tsum += __shfl_xor(tsum, 16, 32);
        Trun = alpha * (Trun + (mrun - mnew) * Zrun) + tsum;
        Zrun = alpha * Zrun + psum;
        mrun = mnew;
        // rescale ctx accumulators (lane-local: lane owns one query column)
#pragma unroll
        for (int t = 0; t < 4; ++t)
#pragma unroll
            for (int j = 0; j < 8; ++j) cacc[t][j] *= alpha;
        // P D-tiles -> B fragment (32x16, keys x queries): register-local
        v16h pb;
#pragma unroll
        for (int j = 0; j < 8; ++j) { pb[j] = (_Float16)p0[j]; pb[8 + j] = (_Float16)p1[j]; }
        // ctx^T += V^T(16hd x 32k) · P(32k x 16q), 4 hd tiles; V^T is
        // pre-transposed f16 so each A-fragment is two b128 loads.
#pragma unroll
        for (int t = 0; t < 4; ++t) {
            v16h av = load_frag_h(VTb, MK, t * 16, kc, lane);
            cacc[t] = __builtin_amdgcn_wmma_f32_16x16x32_f16(
                false, av, false, pb, (short)0, cacc[t], false, false);
        }
    }

    float invZ = 1.0f / Zrun;
    int qcol = qbase + (lane & 15);
    int hd0  = (lane >> 4) << 3;
#pragma unroll
    for (int t = 0; t < 4; ++t)
#pragma unroll
        for (int j = 0; j < 8; ++j) {
            int hd = t * 16 + hd0 + j;
            ctx[(size_t)(b * LQ + qcol) * DIM + h * HDIM + hd] = cacc[t][j] * invZ;
        }
    if (lane < 16) {
        float ent_nat = logf(Zrun) - Trun * invZ;   // -sum p ln p
        ent_bh[(size_t)(b * NHEAD + h) * LQ + qbase + lane] = ent_nat;
    }
}

// mean over heads, normalize by ln(Lk); emit ent output + confidence
__global__ void ent_reduce_kernel(const float* __restrict__ ent_bh,
                                  float* __restrict__ ent_out,
                                  float* __restrict__ conf) {
    int idx = blockIdx.x * blockDim.x + threadIdx.x;
    if (idx >= MQ) return;
    int b = idx / LQ, q = idx % LQ;
    float s = 0.0f;
#pragma unroll
    for (int h = 0; h < NHEAD; ++h) s += ent_bh[(size_t)(b * NHEAD + h) * LQ + q];
    float ent = (s / (float)NHEAD) * (1.0f / logf((float)LK));
    ent_out[idx] = ent;
    conf[idx] = fminf(fmaxf(1.0f - ent, 0.0f), 1.0f);
}

// g[m] = sigmoid( dot(Hh[m], gw2) + gb2 ) * conf[m]   (one wave per row)
__global__ void gate_dot_kernel(const float* __restrict__ Hh,
                                const float* __restrict__ gw2,
                                const float* __restrict__ gb2,
                                const float* __restrict__ conf,
                                float* __restrict__ gbuf) {
    int lane = threadIdx.x & 31;
    int wid  = blockIdx.x * (blockDim.x >> 5) + (threadIdx.x >> 5);
    if (wid >= MQ) return;
    const float* row = Hh + (size_t)wid * DIM;
    float s = 0.0f;
    for (int i = lane; i < DIM; i += 32) s += row[i] * gw2[i];
#pragma unroll
    for (int off = 16; off >= 1; off >>= 1) s += __shfl_xor(s, off, 32);
    if (lane == 0) {
        float z = s + gb2[0];
        gbuf[wid] = (1.0f / (1.0f + __expf(-z))) * conf[wid];
    }
}

// out[m] = LayerNorm( x[m] + r[m] * (gscale ? gscale[m] : 1) )  (wave per row)
__global__ void residual_ln_kernel(const float* __restrict__ x,
                                   const float* __restrict__ r,
                                   const float* __restrict__ gscale,
                                   const float* __restrict__ g,
                                   const float* __restrict__ beta,
                                   float* __restrict__ out) {
    int lane = threadIdx.x & 31;
    int wid  = blockIdx.x * (blockDim.x >> 5) + (threadIdx.x >> 5);
    if (wid >= MQ) return;
    float gs = gscale ? gscale[wid] : 1.0f;
    const float* xr = x + (size_t)wid * DIM;
    const float* rr = r + (size_t)wid * DIM;
    float vbuf[DIM / 32];
    float s = 0.0f, s2 = 0.0f;
#pragma unroll
    for (int i = 0; i < DIM / 32; ++i) {
        int c = lane + i * 32;
        float v = xr[c] + rr[c] * gs;
        vbuf[i] = v; s += v; s2 += v * v;
    }
#pragma unroll
    for (int off = 16; off >= 1; off >>= 1) {
        s  += __shfl_xor(s, off, 32);
        s2 += __shfl_xor(s2, off, 32);
    }
    float mu   = s * (1.0f / DIM);
    float var  = s2 * (1.0f / DIM) - mu * mu;
    float rstd = rsqrtf(var + 1e-5f);
#pragma unroll
    for (int i = 0; i < DIM / 32; ++i) {
        int c = lane + i * 32;
        out[(size_t)wid * DIM + c] = (vbuf[i] - mu) * rstd * g[c] + beta[c];
    }
}

// ---------------------------------------------------------------------------
extern "C" void kernel_launch(void* const* d_in, const int* in_sizes, int n_in,
                              void* d_out, int out_size, void* d_ws, size_t ws_size,
                              hipStream_t stream) {
    const float*         q       = (const float*)d_in[0];
    const float*         kv      = (const float*)d_in[2];
    const unsigned char* kv_mask = (const unsigned char*)d_in[3];
    const float* in_w  = (const float*)d_in[4];
    const float* in_b  = (const float*)d_in[5];
    const float* out_w = (const float*)d_in[6];
    const float* out_b = (const float*)d_in[7];
    const float* ln_g  = (const float*)d_in[8];
    const float* ln_b  = (const float*)d_in[9];
    const float* w1    = (const float*)d_in[10];
    const float* b1    = (const float*)d_in[11];
    const float* w2    = (const float*)d_in[12];
    const float* b2    = (const float*)d_in[13];
    const float* gw1   = (const float*)d_in[14];
    const float* gb1   = (const float*)d_in[15];
    const float* gw2   = (const float*)d_in[16];
    const float* gb2   = (const float*)d_in[17];
    float* outf = (float*)d_out;
    float* ws   = (float*)d_ws;

    // workspace layout (float-sized slots; qp/kp/vp hold f16 in-place), w/ reuse
    const size_t off_qp  = 0;                              // 4096*1024
    const size_t off_kp  = off_qp  + (size_t)MQ * DIM;     // 8192*1024
    const size_t off_vp  = off_kp  + (size_t)MK * DIM;     // 8192*1024
    const size_t off_ctx = off_vp  + (size_t)MK * DIM;     // 4096*1024
    const size_t off_ao  = off_ctx + (size_t)MQ * DIM;     // 4096*1024
    const size_t off_ent = off_ao  + (size_t)MQ * DIM;     // 8*16*512
    const size_t off_cf  = off_ent + (size_t)BATCH * NHEAD * LQ;
    const size_t off_g   = off_cf  + (size_t)MQ;
    _Float16* qp_h  = (_Float16*)(ws + off_qp);
    _Float16* kp_h  = (_Float16*)(ws + off_kp);
    _Float16* vpT_h = (_Float16*)(ws + off_vp);   // [1024 features][MK rows]
    float* ctx     = ws + off_ctx;
    float* ao      = ws + off_ao;
    float* ent_bh  = ws + off_ent;
    float* conf    = ws + off_cf;
    float* gbuf    = ws + off_g;
    float* Hh      = ws + off_qp;   // reuse qp after attention
    float* q2      = ws + off_kp;   // reuse kp
    float* ffn1    = ws + off_vp;   // reuse vp+ctx+ao span (4096*4096)
    float* ffn_out = ws + off_qp;   // reuse qp/Hh

    const int TB = 256;                  // 8 waves / block
    auto blocks = [](int waves) { return (waves + 7) / 8; };

    // 1-3: QKV projections (32x64 tiles); outputs f16 (V transposed)
    gemm_wmma_kernel<false, OUT_F16><<<blocks((MQ/32)*(DIM/64)), TB, 0, stream>>>(
        q, DIM, q, DIM, DIM, MQ, DIM, DIM, in_w, in_b, qp_h, 0);
    gemm_wmma_kernel<false, OUT_F16><<<blocks((MK/32)*(DIM/64)), TB, 0, stream>>>(
        kv, DIM, kv, DIM, DIM, MK, DIM, DIM, in_w + (size_t)DIM*DIM, in_b + DIM, kp_h, 0);
    gemm_wmma_kernel<false, OUT_F16T><<<blocks((MK/32)*(DIM/64)), TB, 0, stream>>>(
        kv, DIM, kv, DIM, DIM, MK, DIM, DIM, in_w + (size_t)2*DIM*DIM, in_b + 2*DIM, vpT_h, 0);
    // 4: attention + entropy
    attn_kernel<<<blocks(BATCH * NHEAD * (LQ/16)), TB, 0, stream>>>(
        qp_h, kp_h, vpT_h, kv_mask, ctx, ent_bh);
    // 5: entropy head-mean -> out tail + conf
    ent_reduce_kernel<<<(MQ + TB - 1)/TB, TB, 0, stream>>>(
        ent_bh, outf + (size_t)MQ * DIM, conf);
    // 6: out projection
    gemm_wmma_kernel<false, OUT_F32><<<blocks((MQ/32)*(DIM/64)), TB, 0, stream>>>(
        ctx, DIM, ctx, DIM, DIM, MQ, DIM, DIM, out_w, out_b, ao, 0);
    // 7: gate hidden = gelu([q ‖ ao] @ gw1^T + gb1)
    gemm_wmma_kernel<true, OUT_F32><<<blocks((MQ/32)*(DIM/64)), TB, 0, stream>>>(
        q, DIM, ao, DIM, DIM, MQ, DIM, 2*DIM, gw1, gb1, Hh, 1);
    // 8: gate scalar
    gate_dot_kernel<<<blocks(MQ), TB, 0, stream>>>(Hh, gw2, gb2, conf, gbuf);
    // 9: q2 = LN(q + ao * g)
    residual_ln_kernel<<<blocks(MQ), TB, 0, stream>>>(q, ao, gbuf, ln_g, ln_b, q2);
    // 10: ffn1 = gelu(q2 @ w1^T + b1)
    gemm_wmma_kernel<false, OUT_F32><<<blocks((MQ/32)*((4*DIM)/64)), TB, 0, stream>>>(
        q2, DIM, q2, DIM, DIM, MQ, 4*DIM, DIM, w1, b1, ffn1, 1);
    // 11: ffn_out = ffn1 @ w2^T + b2
    gemm_wmma_kernel<false, OUT_F32><<<blocks((MQ/32)*(DIM/64)), TB, 0, stream>>>(
        ffn1, 4*DIM, ffn1, 4*DIM, 4*DIM, MQ, DIM, 4*DIM, w2, b2, ffn_out, 0);
    // 12: out = LN(q2 + ffn_out)
    residual_ln_kernel<<<blocks(MQ), TB, 0, stream>>>(
        q2, ffn_out, nullptr, ln_g, ln_b, outf);
}
